// HLMPNN_21225728376839
// MI455X (gfx1250) — matrix operand order repaired and checked by reading, not compiled, not run on hardware
//
#include <hip/hip_runtime.h>
#include <stdint.h>

// ---------------------------------------------------------------------------
// HLMPNN for MI455X (gfx1250, wave32, WMMA).
//
// Roofline: per-node message trick reduces GEMM work from ~1.25 TFLOP to
// ~200 GFLOP; remaining cost is the edge scatter (~1.7 GB/layer, node_msg
// fits in 192MB L2). GEMMs use v_wmma_f32_16x16x32_bf16 (fp32->bf16 via
// native v_cvt_pk_bf16_f32, fp32 accumulate). Scatter/LayerNorm stay fp32.
// ---------------------------------------------------------------------------

#define HID    256
#define MSGH   128
#define INCH   128

#define BM 128
#define BN 128
#define BK 32
#define LDA 40   // bf16 elems per LDS row (32 + 8 pad), 80B => 16B-aligned frags
#define LDB 40

typedef __attribute__((ext_vector_type(16))) __bf16 v16bf;
typedef __attribute__((ext_vector_type(8)))  float  v8f;

union FragAB { v16bf v; uint32_t u[8]; uint4 q[2]; };
union FragC  { v8f   v; float    f[8]; };

// float pair -> packed bf16 dword; plain casts lower to v_cvt_pk_bf16_f32 (RNE)
__device__ __forceinline__ uint32_t pack2_bf16(float lo, float hi) {
  union { uint32_t u; __bf16 h[2]; } p;
  p.h[0] = (__bf16)lo;
  p.h[1] = (__bf16)hi;
  return p.u;
}

// Y[M x Nc] = act(A[M x K] @ W[K x Nc] + bias),  K % 32 == 0, Nc % 128 == 0
template <bool RELU>
__global__ __launch_bounds__(256)
void gemm_bias_kernel(const float* __restrict__ A, const float* __restrict__ W,
                      const float* __restrict__ bias, float* __restrict__ Y,
                      int M, int K, int Nc) {
  __shared__ __bf16 sA[BM * LDA];
  __shared__ __bf16 sB[BN * LDB];

  const int tid   = threadIdx.x;
  const int lane  = tid & 31;
  const int wave  = tid >> 5;
  const int mBase = blockIdx.y * BM;
  const int nBase = blockIdx.x * BN;

  FragC acc[8];
#pragma unroll
  for (int t = 0; t < 8; ++t)
#pragma unroll
    for (int j = 0; j < 8; ++j) acc[t].f[j] = 0.0f;

  for (int k0 = 0; k0 < K; k0 += BK) {
    // ---- stage A tile (BM x BK fp32 -> bf16), 4 float4 per thread ----
#pragma unroll
    for (int it = 0; it < 4; ++it) {
      int s   = tid + it * 256;        // 1024 float4 slots
      int row = s >> 3;                // BK/4 = 8 slots per row
      int kq  = (s & 7) << 2;
      float4 v = make_float4(0.f, 0.f, 0.f, 0.f);
      int gr = mBase + row;
      if (gr < M) v = *(const float4*)(A + (size_t)gr * K + k0 + kq);
      uint32_t p0 = pack2_bf16(v.x, v.y);
      uint32_t p1 = pack2_bf16(v.z, v.w);
      *(uint2*)(&sA[row * LDA + kq]) = make_uint2(p0, p1);
    }
    // ---- stage B tile transposed: W[k][n] -> sB[n][k] (bf16) ----
#pragma unroll
    for (int it = 0; it < 4; ++it) {
      int s  = tid + it * 256;         // 1024 float4 slots
      int kk = s >> 5;                 // BN/4 = 32 slots per k-row
      int nq = (s & 31) << 2;
      float4 w4 = *(const float4*)(W + (size_t)(k0 + kk) * Nc + nBase + nq);
      sB[(nq + 0) * LDB + kk] = (__bf16)w4.x;
      sB[(nq + 1) * LDB + kk] = (__bf16)w4.y;
      sB[(nq + 2) * LDB + kk] = (__bf16)w4.z;
      sB[(nq + 3) * LDB + kk] = (__bf16)w4.w;
    }
    __syncthreads();

    // ---- A fragment: row = wave*16 + (lane&15)
    // lanes 0-15: K 0-7 / 16-23 ; lanes 16-31: K 8-15 / 24-31  (ISA 7.12.2)
    FragAB a;
    {
      int row = (wave << 4) + (lane & 15);
      int kb  = (lane >> 4) << 3;      // 0 or 8
      const __bf16* pa = &sA[row * LDA];
      a.q[0] = *(const uint4*)(pa + kb);
      a.q[1] = *(const uint4*)(pa + kb + 16);
    }
#pragma unroll
    for (int t = 0; t < 8; ++t) {
      // B fragment: col = t*16 + (lane&15); lanes 0-15: K 0-7 / 16-23, 16-31: K 8-15 / 24-31
      FragAB b;
      int col = (t << 4) + (lane & 15);
      int kb  = (lane >> 4) << 3;      // 0 or 8
      const __bf16* pb = &sB[col * LDB + kb];
      b.q[0] = *(const uint4*)(pb);
      b.q[1] = *(const uint4*)(pb + 16);
      acc[t].v = __builtin_amdgcn_wmma_f32_16x16x32_bf16(
          false, a.v, false, b.v, (short)0, acc[t].v, false, false);
    }
    __syncthreads();
  }

  // ---- store: C/D layout: N = lane&15, M = r + 8*(lane>=16) ----
#pragma unroll
  for (int t = 0; t < 8; ++t) {
    int col = nBase + (t << 4) + (lane & 15);
    float bv = bias[col];
    int rAdd = (lane >> 4) << 3;       // 0 or 8
#pragma unroll
    for (int r = 0; r < 8; ++r) {
      int grow = mBase + (wave << 4) + rAdd + r;
      if (grow < M) {
        float v = acc[t].f[r] + bv;
        if (RELU) v = fmaxf(v, 0.0f);
        Y[(size_t)grow * Nc + col] = v;
      }
    }
  }
}

// ---------------- graph / elementwise kernels ----------------

__global__ void softmax_beta_kernel(const float* __restrict__ beta,
                                    float* __restrict__ betas, int n) {
  if (blockIdx.x == 0 && threadIdx.x == 0) {
    float mx = -3.0e38f;
    for (int i = 0; i < n; ++i) mx = fmaxf(mx, beta[i]);
    float s = 0.f, tmp[16];
    for (int i = 0; i < n; ++i) { tmp[i] = __expf(beta[i] - mx); s += tmp[i]; }
    float inv = 1.0f / s;
    for (int i = 0; i < n; ++i) betas[i] = tmp[i] * inv;
  }
}

__global__ void counts_init_kernel(float* __restrict__ counts, int n) {
  int i = blockIdx.x * blockDim.x + threadIdx.x;
  if (i < n) counts[i] = 1.0f;                  // self loop
}

__global__ void counts_scatter_kernel(const int* __restrict__ dst,
                                      float* __restrict__ counts, int E) {
  int e = blockIdx.x * blockDim.x + threadIdx.x;
  if (e < E) atomicAdd(&counts[dst[e]], 1.0f);
}

// one wave per edge; lane handles 8 columns (2x float4 gather + 8 f32 atomics)
__global__ __launch_bounds__(256)
void scatter_msg_kernel(const float* __restrict__ nm, const int* __restrict__ src,
                        const int* __restrict__ dst, float* __restrict__ m, int E) {
  long long t = (long long)blockIdx.x * blockDim.x + threadIdx.x;
  if (t >= (long long)E * 32) return;
  int e  = (int)(t >> 5);
  int c0 = ((int)t & 31) << 3;
  int s = src[e], d = dst[e];
  const float4* ps = (const float4*)(nm + (size_t)s * HID + c0);
  float4 v0 = ps[0], v1 = ps[1];
  float* pd = m + (size_t)d * HID + c0;
  atomicAdd(pd + 0, v0.x); atomicAdd(pd + 1, v0.y);
  atomicAdd(pd + 2, v0.z); atomicAdd(pd + 3, v0.w);
  atomicAdd(pd + 4, v1.x); atomicAdd(pd + 5, v1.y);
  atomicAdd(pd + 6, v1.z); atomicAdd(pd + 7, v1.w);
}

// h = z + m / counts   (in place into m)
__global__ void combine_kernel(const float* __restrict__ z, float* __restrict__ m,
                               const float* __restrict__ counts, long long total) {
  long long t = (long long)blockIdx.x * blockDim.x + threadIdx.x;
  if (t >= total) return;
  int row = (int)(t >> 8);
  m[t] = z[t] + m[t] / counts[row];
}

// out = betas[0] * z
__global__ void init_out_kernel(const float* __restrict__ z,
                                const float* __restrict__ betas,
                                float* __restrict__ out, long long total) {
  long long t = (long long)blockIdx.x * blockDim.x + threadIdx.x;
  if (t >= total) return;
  out[t] = betas[0] * z[t];
}

// rowwise LayerNorm (in place on z) + out += betas[idx] * z ; one wave per row
__global__ __launch_bounds__(256)
void layernorm_acc_kernel(float* __restrict__ z, const float* __restrict__ g,
                          const float* __restrict__ b, const float* __restrict__ betas,
                          int betaIdx, float* __restrict__ out, int n) {
  int lane = threadIdx.x & 31;
  int wave = threadIdx.x >> 5;
  int row  = blockIdx.x * 8 + wave;
  if (row >= n) return;
  float* pz = z + (size_t)row * HID;
  int c0 = lane * 8;
  float4 a0 = *(const float4*)(pz + c0);
  float4 a1 = *(const float4*)(pz + c0 + 4);
  float vals[8] = {a0.x, a0.y, a0.z, a0.w, a1.x, a1.y, a1.z, a1.w};
  float s = 0.f, ss = 0.f;
#pragma unroll
  for (int i = 0; i < 8; ++i) { s += vals[i]; ss = fmaf(vals[i], vals[i], ss); }
#pragma unroll
  for (int off = 16; off > 0; off >>= 1) {
    s  += __shfl_xor(s, off, 32);
    ss += __shfl_xor(ss, off, 32);
  }
  float mu  = s * (1.0f / HID);
  float var = ss * (1.0f / HID) - mu * mu;
  float inv = rsqrtf(var + 1e-5f);
  float bw  = betas[betaIdx];
  float* po = out + (size_t)row * HID;
#pragma unroll
  for (int i = 0; i < 8; ++i) {
    int c = c0 + i;
    float y = (vals[i] - mu) * inv * g[c] + b[c];
    pz[c] = y;
    po[c] += bw * y;
  }
}

// ---------------------------------------------------------------------------

extern "C" void kernel_launch(void* const* d_in, const int* in_sizes, int n_in,
                              void* d_out, int out_size, void* d_ws, size_t ws_size,
                              hipStream_t stream) {
  const float* x    = (const float*)d_in[0];
  const int*   ei   = (const int*)  d_in[1];
  const float* Win  = (const float*)d_in[2];
  const float* binp = (const float*)d_in[3];
  const float* W1   = (const float*)d_in[4];
  const float* b1   = (const float*)d_in[5];
  const float* W2   = (const float*)d_in[6];
  const float* b2   = (const float*)d_in[7];
  const float* U1   = (const float*)d_in[8];
  const float* c1   = (const float*)d_in[9];
  const float* U2   = (const float*)d_in[10];
  const float* c2   = (const float*)d_in[11];
  const float* lng  = (const float*)d_in[12];
  const float* lnb  = (const float*)d_in[13];
  const float* beta = (const float*)d_in[14];

  const int N = in_sizes[0] / INCH;
  const int E = in_sizes[1] / 2;
  const int L = in_sizes[4] / (HID * MSGH);
  const int* src = ei;
  const int* dst = ei + E;

  const size_t nh = (size_t)N * HID;

  // workspace layout (floats): betas | counts | z | t1 | nm(->o1) | m(->h)
  float* ws     = (float*)d_ws;
  float* betas  = ws;                                  // 16
  float* counts = ws + 16;                             // N
  float* z      = counts + (((size_t)N + 63) / 64) * 64;
  float* t1     = z  + nh;                             // N * MSGH
  float* nm     = t1 + (size_t)N * MSGH;               // N * HID
  float* m      = nm + nh;                             // N * HID
  // total ~180 MB

  const int  mb  = (N + BM - 1) / BM;
  const dim3 blk(256);
  const long long totNH = (long long)nh;
  const unsigned gNH = (unsigned)((totNH + 255) / 256);

  softmax_beta_kernel<<<1, 32, 0, stream>>>(beta, betas, L + 1);
  counts_init_kernel<<<(N + 255) / 256, 256, 0, stream>>>(counts, N);
  counts_scatter_kernel<<<(E + 255) / 256, 256, 0, stream>>>(dst, counts, E);

  // z0 = x @ Win + bin
  gemm_bias_kernel<false><<<dim3(HID / BN, mb), blk, 0, stream>>>(
      x, Win, binp, z, N, INCH, HID);
  init_out_kernel<<<gNH, 256, 0, stream>>>(z, betas, (float*)d_out, totNH);

  for (int l = 0; l < L; ++l) {
    // per-NODE message MLP (edge messages depend only on src node)
    gemm_bias_kernel<true><<<dim3(MSGH / BN, mb), blk, 0, stream>>>(
        z, W1 + (size_t)l * HID * MSGH, b1 + (size_t)l * MSGH, t1, N, HID, MSGH);
    gemm_bias_kernel<false><<<dim3(HID / BN, mb), blk, 0, stream>>>(
        t1, W2 + (size_t)l * MSGH * HID, b2 + (size_t)l * HID, nm, N, MSGH, HID);

    // m = nm (self loop) + scatter nm[src] -> m[dst]
    hipMemcpyAsync(m, nm, nh * sizeof(float), hipMemcpyDeviceToDevice, stream);
    long long totE = (long long)E * 32;
    scatter_msg_kernel<<<(unsigned)((totE + 255) / 256), 256, 0, stream>>>(
        nm, src, dst, m, E);

    // h = z + m / counts   (into m)
    combine_kernel<<<gNH, 256, 0, stream>>>(z, m, counts, totNH);

    // node update MLP: o1 = relu(h@U1+c1) (into nm), z = o1@U2+c2
    gemm_bias_kernel<true><<<dim3(HID / BN, mb), blk, 0, stream>>>(
        m, U1 + (size_t)l * HID * HID, c1 + (size_t)l * HID, nm, N, HID, HID);
    gemm_bias_kernel<false><<<dim3(HID / BN, mb), blk, 0, stream>>>(
        nm, U2 + (size_t)l * HID * HID, c2 + (size_t)l * HID, z, N, HID, HID);

    // LayerNorm in place + out += betas[l+1] * z
    layernorm_acc_kernel<<<(N + 7) / 8, 256, 0, stream>>>(
        z, lng + (size_t)l * HID, lnb + (size_t)l * HID, betas, l + 1,
        (float*)d_out, N);
  }
}
